// FakeLvqLinear_68590627717602
// MI455X (gfx1250) — matrix operand order repaired and checked by reading, not compiled, not run on hardware
//
#include <hip/hip_runtime.h>
#include <stdint.h>

// ---------------------------------------------------------------------------
// FakeLvqLinear for MI455X (gfx1250, wave32, WMMA)
//   out[M,N] = x[M,K] @ W[N,K]^T + bias,  W reconstructed from additive LUTs.
//   M=8192, N=OUT=4096, K=IN=4096 -> 275 GFLOP GEMM, compute bound.
//
//   Fast path (ws >= 192 MB): pre-split x and reconstructed W into bf16
//   hi/lo planes once (bf16x3 split => fp32-class accuracy at bf16 WMMA
//   rate). GEMM hot loop: double-buffered LDS (64 KB, XOR-swizzled,
//   conflict-free b128 fragment loads), register prefetch of the next tile,
//   ONE barrier per k-step, 24 v_wmma_f32_16x16x32_bf16 per wave per step.
//   Fallback (ws >= 64 MB): fp32 W in ws, split done in-loop.
// ---------------------------------------------------------------------------

typedef __attribute__((ext_vector_type(16))) __bf16 bf16x16;
typedef __attribute__((ext_vector_type(8)))  float  f32x8;

#define IN_DIM   4096
#define OUT_DIM  4096
#define VEC      4
#define V_DIM    (IN_DIM / VEC)     // 1024
#define LUT_SIZE 256
#define GROUPS   (IN_DIM / 128)     // 32
#define KP       (IN_DIM / 2)       // packed bf16 pairs per row: 2048
#define T_STEPS  (KP / 16)          // 128 k-steps of 32 elements

#define BM 128
#define BN 128
#define BK 32

#define ARR   (BM * 16)             // uint32 per plane-tile (no pad): 2048
#define STAGE (4 * ARR)             // uint32 per stage: 8192 (32 KB)

static __device__ __forceinline__ uint16_t f2bf(float f) {
  uint32_t u = __float_as_uint(f);
  u += 0x7FFFu + ((u >> 16) & 1u);          // round-to-nearest-even
  return (uint16_t)(u >> 16);
}
static __device__ __forceinline__ float bf2f(uint16_t h) {
  return __uint_as_float(((uint32_t)h) << 16);
}
static __device__ __forceinline__ f32x8 wmma_bf16(bf16x16 a, bf16x16 b, f32x8 c) {
  return __builtin_amdgcn_wmma_f32_16x16x32_bf16(
      /*neg_a=*/false, a, /*neg_b=*/false, b,
      /*c_mod=*/(short)0, c, /*reuse_a=*/false, /*reuse_b=*/false);
}

union Frag { uint32_t u[8]; bf16x16 v; };

// XOR-swizzled LDS offset (uint32 units) of 16B chunk `chunk` in row `row`.
// Fragment reads touch 16 consecutive rows: (row&3, (row>>2)&3) is a bijection
// over the 16 4-bank spans -> conflict-free ds_load_b128.
static __device__ __forceinline__ int swz(int row, int chunk) {
  return row * 16 + ((chunk ^ ((row >> 2) & 3)) << 2);
}

// Split 4 consecutive fp32 K-values into hi/lo packed-bf16 dword pairs.
static __device__ __forceinline__ void split4(float4 x, uint32_t* hi, uint32_t* lo) {
  uint16_t h0 = f2bf(x.x), h1 = f2bf(x.y), h2 = f2bf(x.z), h3 = f2bf(x.w);
  hi[0] = (uint32_t)h0 | ((uint32_t)h1 << 16);
  hi[1] = (uint32_t)h2 | ((uint32_t)h3 << 16);
  uint16_t l0 = f2bf(x.x - bf2f(h0)), l1 = f2bf(x.y - bf2f(h1));
  uint16_t l2 = f2bf(x.z - bf2f(h2)), l3 = f2bf(x.w - bf2f(h3));
  lo[0] = (uint32_t)l0 | ((uint32_t)l1 << 16);
  lo[1] = (uint32_t)l2 | ((uint32_t)l3 << 16);
}

// Gather one weight float4 for (v, o): sum of two LUT codewords * group scale
static __device__ __forceinline__ float4 lvq_gather(
    int v, int o, const int* __restrict__ index, const float* __restrict__ luts,
    const float* __restrict__ scales) {
  int i0 = index[(size_t)v * OUT_DIM + o];
  int i1 = index[(size_t)(V_DIM + v) * OUT_DIM + o];
  const float4* t0 = (const float4*)luts + (size_t)v * LUT_SIZE;
  const float4* t1 = (const float4*)luts + (size_t)(V_DIM + v) * LUT_SIZE;
  float4 a = t0[i0];
  float4 b = t1[i1];
  float s = scales[(size_t)o * GROUPS + (v >> 5)];   // (v*4)/128 == v/32
  float4 w;
  w.x = (a.x + b.x) * s;
  w.y = (a.y + b.y) * s;
  w.z = (a.z + b.z) * s;
  w.w = (a.w + b.w) * s;
  return w;
}

// ========================= FAST PATH (bf16 planes) =========================

__global__ __launch_bounds__(256) void lvq_reconstruct_planes(
    const int*   __restrict__ index,   // [2, V, OUT]
    const float* __restrict__ luts,    // [2, V, 256, 4]
    const float* __restrict__ scales,  // [OUT, GROUPS]
    uint32_t*    __restrict__ Wh,      // [OUT, K/2]
    uint32_t*    __restrict__ Wl)      // [OUT, K/2]
{
  int gid = blockIdx.x * 256 + threadIdx.x;   // gid = v * OUT + o, exact grid
  int v = gid >> 12;                          // / OUT_DIM
  int o = gid & (OUT_DIM - 1);
  float4 w = lvq_gather(v, o, index, luts, scales);
  uint32_t h[2], l[2];
  split4(w, h, l);
  uint2 hv; hv.x = h[0]; hv.y = h[1];
  uint2 lv; lv.x = l[0]; lv.y = l[1];
  ((uint2*)(Wh + (size_t)o * KP))[v] = hv;
  ((uint2*)(Wl + (size_t)o * KP))[v] = lv;
}

__global__ __launch_bounds__(256) void lvq_split_x(
    const float* __restrict__ X,       // [M, K]
    uint32_t*    __restrict__ Xh,      // [M, K/2]
    uint32_t*    __restrict__ Xl)      // [M, K/2]
{
  int gid = blockIdx.x * 256 + threadIdx.x;   // over M*K/4 float4 chunks
  int row = gid >> 10;                        // / (K/4)
  int c   = gid & 1023;
  float4 xv = ((const float4*)(X + (size_t)row * IN_DIM))[c];
  uint32_t h[2], l[2];
  split4(xv, h, l);
  uint2 hv; hv.x = h[0]; hv.y = h[1];
  uint2 lv; lv.x = l[0]; lv.y = l[1];
  ((uint2*)(Xh + (size_t)row * KP))[c] = hv;
  ((uint2*)(Xl + (size_t)row * KP))[c] = lv;
}

// per-thread tile prefetch: 2 chunks (uint4) from each of the 4 planes
static __device__ __forceinline__ void tile_load(
    const uint32_t* pXh, const uint32_t* pXl,
    const uint32_t* pWh, const uint32_t* pWl,
    int kp0, int cbase, uint4* pf) {
  #pragma unroll
  for (int i = 0; i < 2; ++i) {
    int c4 = (cbase + i) * 4;
    pf[0 + i] = *(const uint4*)(pXh + kp0 + c4);
    pf[2 + i] = *(const uint4*)(pXl + kp0 + c4);
    pf[4 + i] = *(const uint4*)(pWh + kp0 + c4);
    pf[6 + i] = *(const uint4*)(pWl + kp0 + c4);
  }
}

static __device__ __forceinline__ void tile_store(
    uint32_t* st, int crow, int cbase, const uint4* pf) {
  #pragma unroll
  for (int i = 0; i < 2; ++i) {
    int c = cbase + i;
    *(uint4*)(st + 0 * ARR + swz(crow, c)) = pf[0 + i];
    *(uint4*)(st + 1 * ARR + swz(crow, c)) = pf[2 + i];
    *(uint4*)(st + 2 * ARR + swz(crow, c)) = pf[4 + i];
    *(uint4*)(st + 3 * ARR + swz(crow, c)) = pf[6 + i];
  }
}

__global__ __launch_bounds__(256) void lvq_gemm_planes(
    const uint32_t* __restrict__ Xh, const uint32_t* __restrict__ Xl,
    const uint32_t* __restrict__ Wh, const uint32_t* __restrict__ Wl,
    const float* __restrict__ bias, float* __restrict__ out)
{
  __shared__ uint32_t smem[2 * STAGE];   // 64 KB double-buffered

  const int tid  = threadIdx.x;
  const int lane = tid & 31;
  const int wave = tid >> 5;
  const int wm   = wave & 1;       // 2 waves along M (64 rows each)
  const int wn   = wave >> 1;      // 4 waves along N (32 cols each)
  const int l15  = lane & 15;
  const int hh   = lane >> 4;      // half-wave id (0/1)

  const int m_base = blockIdx.y * BM;
  const int n_base = blockIdx.x * BN;

  // copy roles: 512 chunks per plane-tile, 2 per thread
  const int crow  = tid >> 1;           // 0..127
  const int cbase = (tid & 1) * 2;      // chunk 0/1 or 2/3

  const uint32_t* pXh = Xh + (size_t)(m_base + crow) * KP;
  const uint32_t* pXl = Xl + (size_t)(m_base + crow) * KP;
  const uint32_t* pWh = Wh + (size_t)(n_base + crow) * KP;
  const uint32_t* pWl = Wl + (size_t)(n_base + crow) * KP;

  f32x8 acc[4][2] = {};
  uint4 pf[8];

  // prologue: tile 0 -> stage 0
  tile_load(pXh, pXl, pWh, pWl, 0, cbase, pf);
  tile_store(smem, crow, cbase, pf);
  __syncthreads();

  for (int kt = 0; kt < T_STEPS; ++kt) {
    const uint32_t* st = smem + (kt & 1) * STAGE;
    const uint32_t* stAh = st + 0 * ARR;
    const uint32_t* stAl = st + 1 * ARR;
    const uint32_t* stBh = st + 2 * ARR;
    const uint32_t* stBl = st + 3 * ARR;

    // issue next tile's global loads before the WMMA chain (latency hiding)
    if (kt + 1 < T_STEPS)
      tile_load(pXh, pXl, pWh, pWl, (kt + 1) * 16, cbase, pf);

    // ---- B fragments (kept live across the mt loop) ----
    Frag bh[2], bl[2];
    #pragma unroll
    for (int nt = 0; nt < 2; ++nt) {
      int row = wn * 32 + nt * 16 + l15;
      *(uint4*)&bh[nt].u[0] = *(const uint4*)(stBh + swz(row, 2 * hh + 0));
      *(uint4*)&bh[nt].u[4] = *(const uint4*)(stBh + swz(row, 2 * hh + 1));
      *(uint4*)&bl[nt].u[0] = *(const uint4*)(stBl + swz(row, 2 * hh + 0));
      *(uint4*)&bl[nt].u[4] = *(const uint4*)(stBl + swz(row, 2 * hh + 1));
    }

    // ---- A fragments per M-subtile, then 3 WMMAs per C tile ----
    #pragma unroll
    for (int mt = 0; mt < 4; ++mt) {
      int row = wm * 64 + mt * 16 + l15;
      Frag ah, al;
      *(uint4*)&ah.u[0] = *(const uint4*)(stAh + swz(row, 2 * hh + 0));
      *(uint4*)&ah.u[4] = *(const uint4*)(stAh + swz(row, 2 * hh + 1));
      *(uint4*)&al.u[0] = *(const uint4*)(stAl + swz(row, 2 * hh + 0));
      *(uint4*)&al.u[4] = *(const uint4*)(stAl + swz(row, 2 * hh + 1));
      #pragma unroll
      for (int nt = 0; nt < 2; ++nt) {
        acc[mt][nt] = wmma_bf16(ah.v, bh[nt].v, acc[mt][nt]);
        acc[mt][nt] = wmma_bf16(ah.v, bl[nt].v, acc[mt][nt]);
        acc[mt][nt] = wmma_bf16(al.v, bh[nt].v, acc[mt][nt]);
      }
    }

    // store prefetched tile into the other stage, single barrier per step
    if (kt + 1 < T_STEPS)
      tile_store(smem + ((kt + 1) & 1) * STAGE, crow, cbase, pf);
    __syncthreads();
  }

  // ---- epilogue: C layout lane<16 -> M=v, lane>=16 -> M=v+8 ----
  #pragma unroll
  for (int nt = 0; nt < 2; ++nt) {
    int col = n_base + wn * 32 + nt * 16 + l15;
    float bv = bias[col];
    #pragma unroll
    for (int mt = 0; mt < 4; ++mt) {
      #pragma unroll
      for (int v = 0; v < 8; ++v) {
        int row = m_base + wm * 64 + mt * 16 + hh * 8 + v;
        out[(size_t)row * OUT_DIM + col] = acc[mt][nt][v] + bv;
      }
    }
  }
}

// ===================== FALLBACK PATH (fp32 W in ws) ========================

#define LDSW 20   // padded rows for the fallback kernel

__global__ __launch_bounds__(256) void lvq_reconstruct_f32(
    const int*   __restrict__ index, const float* __restrict__ luts,
    const float* __restrict__ scales, float* __restrict__ W)
{
  int gid = blockIdx.x * 256 + threadIdx.x;
  int v = gid >> 12;
  int o = gid & (OUT_DIM - 1);
  float4 w = lvq_gather(v, o, index, luts, scales);
  ((float4*)(W + (size_t)o * IN_DIM))[v] = w;
}

__global__ __launch_bounds__(256) void lvq_gemm_f32w(
    const float* __restrict__ X, const float* __restrict__ W,
    const float* __restrict__ bias, float* __restrict__ out)
{
  __shared__ uint32_t smem[4 * BM * LDSW];
  uint32_t* sAh = smem;
  uint32_t* sAl = smem + 1 * BM * LDSW;
  uint32_t* sBh = smem + 2 * BM * LDSW;
  uint32_t* sBl = smem + 3 * BM * LDSW;

  const int tid  = threadIdx.x;
  const int lane = tid & 31;
  const int wave = tid >> 5;
  const int wm   = wave & 1;
  const int wn   = wave >> 1;
  const int l15  = lane & 15;
  const int hh   = lane >> 4;

  const int m_base = blockIdx.y * BM;
  const int n_base = blockIdx.x * BN;

  f32x8 acc[4][2] = {};

  for (int k0 = 0; k0 < IN_DIM; k0 += BK) {
    #pragma unroll
    for (int i = 0; i < 4; ++i) {
      int linear = tid + i * 256;
      int row = linear >> 3;
      int c4  = linear & 7;
      float4 xv = *(const float4*)(X + (size_t)(m_base + row) * IN_DIM + k0 + c4 * 4);
      float4 wv = *(const float4*)(W + (size_t)(n_base + row) * IN_DIM + k0 + c4 * 4);
      uint32_t h[2], l[2];
      split4(xv, h, l);
      sAh[row * LDSW + c4 * 2 + 0] = h[0];
      sAh[row * LDSW + c4 * 2 + 1] = h[1];
      sAl[row * LDSW + c4 * 2 + 0] = l[0];
      sAl[row * LDSW + c4 * 2 + 1] = l[1];
      split4(wv, h, l);
      sBh[row * LDSW + c4 * 2 + 0] = h[0];
      sBh[row * LDSW + c4 * 2 + 1] = h[1];
      sBl[row * LDSW + c4 * 2 + 0] = l[0];
      sBl[row * LDSW + c4 * 2 + 1] = l[1];
    }
    __syncthreads();

    Frag bh[2], bl[2];
    #pragma unroll
    for (int nt = 0; nt < 2; ++nt) {
      const uint32_t* p = sBh + (wn * 32 + nt * 16 + l15) * LDSW + hh * 8;
      *(uint4*)&bh[nt].u[0] = *(const uint4*)(p + 0);
      *(uint4*)&bh[nt].u[4] = *(const uint4*)(p + 4);
      const uint32_t* q = sBl + (wn * 32 + nt * 16 + l15) * LDSW + hh * 8;
      *(uint4*)&bl[nt].u[0] = *(const uint4*)(q + 0);
      *(uint4*)&bl[nt].u[4] = *(const uint4*)(q + 4);
    }
    #pragma unroll
    for (int mt = 0; mt < 4; ++mt) {
      Frag ah, al;
      const uint32_t* p = sAh + (wm * 64 + mt * 16 + l15) * LDSW + hh * 8;
      *(uint4*)&ah.u[0] = *(const uint4*)(p + 0);
      *(uint4*)&ah.u[4] = *(const uint4*)(p + 4);
      const uint32_t* q = sAl + (wm * 64 + mt * 16 + l15) * LDSW + hh * 8;
      *(uint4*)&al.u[0] = *(const uint4*)(q + 0);
      *(uint4*)&al.u[4] = *(const uint4*)(q + 4);
      #pragma unroll
      for (int nt = 0; nt < 2; ++nt) {
        acc[mt][nt] = wmma_bf16(ah.v, bh[nt].v, acc[mt][nt]);
        acc[mt][nt] = wmma_bf16(ah.v, bl[nt].v, acc[mt][nt]);
        acc[mt][nt] = wmma_bf16(al.v, bh[nt].v, acc[mt][nt]);
      }
    }
    __syncthreads();
  }

  #pragma unroll
  for (int nt = 0; nt < 2; ++nt) {
    int col = n_base + wn * 32 + nt * 16 + l15;
    float bv = bias[col];
    #pragma unroll
    for (int mt = 0; mt < 4; ++mt) {
      #pragma unroll
      for (int v = 0; v < 8; ++v) {
        int row = m_base + wm * 64 + mt * 16 + hh * 8 + v;
        out[(size_t)row * OUT_DIM + col] = acc[mt][nt][v] + bv;
      }
    }
  }
}

// ---------------------------------------------------------------------------
extern "C" void kernel_launch(void* const* d_in, const int* in_sizes, int n_in,
                              void* d_out, int out_size, void* d_ws, size_t ws_size,
                              hipStream_t stream) {
  (void)n_in; (void)out_size;
  const float* x      = (const float*)d_in[0];   // [B,S,IN] fp32
  const int*   index  = (const int*)  d_in[1];   // [2,V,OUT] int32
  const float* luts   = (const float*)d_in[2];   // [2,V,256,4] fp32
  const float* scales = (const float*)d_in[3];   // [OUT, GROUPS] fp32
  const float* bias   = (const float*)d_in[4];   // [OUT] fp32
  float*       out    = (float*)d_out;           // [B,S,OUT] fp32

  const size_t M = (size_t)in_sizes[0] / IN_DIM; // 8192
  dim3 grid(OUT_DIM / BN, (unsigned)(M / BM));

  const size_t xplane = M * IN_DIM * 2;                 // bytes per x bf16 plane
  const size_t wplane = (size_t)OUT_DIM * IN_DIM * 2;   // bytes per W bf16 plane
  const size_t need_fast = 2 * xplane + 2 * wplane;     // 192 MB at ref sizes

  if (ws_size >= need_fast) {
    char* ws = (char*)d_ws;
    uint32_t* Xh = (uint32_t*)(ws);
    uint32_t* Xl = (uint32_t*)(ws + xplane);
    uint32_t* Wh = (uint32_t*)(ws + 2 * xplane);
    uint32_t* Wl = (uint32_t*)(ws + 2 * xplane + wplane);

    lvq_reconstruct_planes<<<(V_DIM * OUT_DIM) / 256, 256, 0, stream>>>(
        index, luts, scales, Wh, Wl);
    lvq_split_x<<<(unsigned)(M * IN_DIM / 4 / 256), 256, 0, stream>>>(x, Xh, Xl);
    lvq_gemm_planes<<<grid, dim3(256), 0, stream>>>(Xh, Xl, Wh, Wl, bias, out);
  } else {
    float* W = (float*)d_ws;   // 64 MB fp32 W
    lvq_reconstruct_f32<<<(V_DIM * OUT_DIM) / 256, 256, 0, stream>>>(
        index, luts, scales, W);
    lvq_gemm_f32w<<<grid, dim3(256), 0, stream>>>(x, W, bias, out);
  }
}
